// self_attention_gcn_72353019068846
// MI455X (gfx1250) — matrix operand order repaired and checked by reading, not compiled, not run on hardware
//
#include <hip/hip_runtime.h>
#include <hip/hip_bf16.h>

typedef __attribute__((ext_vector_type(16))) _Float16 v16h;
typedef __attribute__((ext_vector_type(8)))  _Float16 v8h;
typedef __attribute__((ext_vector_type(8)))  float    v8f;

#define N_ 64
#define C_ 64
#define T_ 300
#define V_ 25
#define O_ 64
#define S_ 3
#define E_ 12

static __device__ __forceinline__ v16h cat16(v8h lo, v8h hi) {
  return __builtin_shufflevector(lo, hi, 0,1,2,3,4,5,6,7,8,9,10,11,12,13,14,15);
}
static __device__ __forceinline__ v8f wmma_f16(v16h a, v16h b, v8f c) {
  // D = A(16x32 f16) * B(32x16 f16) + C(16x16 f32)
  return __builtin_amdgcn_wmma_f32_16x16x32_f16(false, a, false, b, (short)0, c, false, false);
}

// ---------------- kernel 0: prep (Wd->f16, BN folding, sum bd) ----------------
__global__ __launch_bounds__(128) void k_prep(const float* __restrict__ Wd,
    const float* __restrict__ bd, const float* __restrict__ gamma,
    const float* __restrict__ beta, const float* __restrict__ mean,
    const float* __restrict__ var,
    _Float16* __restrict__ WdH, float* __restrict__ bnS,
    float* __restrict__ bnB, float* __restrict__ bdS) {
  int idx = blockIdx.x * blockDim.x + threadIdx.x;
  if (idx < S_ * O_ * C_) WdH[idx] = (_Float16)Wd[idx];
  if (idx < O_) {
    float sc = gamma[idx] * rsqrtf(var[idx] + 1e-5f);
    bnS[idx] = sc;
    bnB[idx] = beta[idx] - mean[idx] * sc;
    bdS[idx] = bd[idx] + bd[O_ + idx] + bd[2 * O_ + idx];
  }
}

// ---------------- kernel 1: temporal collapse t1/t2[s,n,c,v] ----------------
__global__ __launch_bounds__(32) void k_tcollapse(const float* __restrict__ x,
    const float* __restrict__ Wt1, const float* __restrict__ bt1,
    const float* __restrict__ Wt2, const float* __restrict__ bt2,
    float* __restrict__ t1o, float* __restrict__ t2o) {
  __shared__ float w[6 * 304];
  const int nc = blockIdx.x;
  const int n = nc >> 6, c = nc & 63;
  const int v = threadIdx.x;
  for (int idx = v; idx < 6 * T_; idx += 32) {
    int j = idx / T_, t = idx - j * T_;
    w[j * 304 + t] = (j < 3) ? Wt1[j * T_ + t] : Wt2[(j - 3) * T_ + t];
  }
  __syncthreads();
  const float* xp = x + (size_t)(n * C_ + c) * T_ * V_;
  float acc[6] = {0.f, 0.f, 0.f, 0.f, 0.f, 0.f};
  for (int t = 0; t < T_; ++t) {
    float xv = (v < V_) ? xp[t * V_ + v] : 0.f;
#pragma unroll
    for (int j = 0; j < 6; ++j) acc[j] += xv * w[j * 304 + t];
  }
  if (v < V_) {
#pragma unroll
    for (int s = 0; s < 3; ++s) {
      t1o[(((size_t)s * N_ + n) * C_ + c) * V_ + v] = acc[s] + bt1[s];
      t2o[(((size_t)s * N_ + n) * C_ + c) * V_ + v] = acc[3 + s] + bt2[s];
    }
  }
}

// ---------------- kernel 2: q/k, logits, softmax -> A1T[s,n,u(32),v(32)] f16 ----------------
__global__ __launch_bounds__(128) void k_attn(const float* __restrict__ t1i,
    const float* __restrict__ t2i,
    const float* __restrict__ Wq, const float* __restrict__ bq,
    const float* __restrict__ Wk, const float* __restrict__ bk,
    _Float16* __restrict__ A1T) {
  __shared__ float t1L[C_ * V_], t2L[C_ * V_];
  __shared__ float qL[E_ * V_], kL[E_ * V_];
  __shared__ float lg[V_ * V_];  // [u][v]
  const int sn = blockIdx.x;
  const int s = sn >> 6;
  const int tid = threadIdx.x;
  for (int idx = tid; idx < C_ * V_; idx += 128) {
    t1L[idx] = t1i[(size_t)sn * C_ * V_ + idx];
    t2L[idx] = t2i[(size_t)sn * C_ * V_ + idx];
  }
  __syncthreads();
  for (int idx = tid; idx < E_ * V_; idx += 128) {
    int e = idx / V_, v = idx - e * V_;
    float sq = bq[s * E_ + e], sk = bk[s * E_ + e];
    for (int c = 0; c < C_; ++c) {
      sq += Wq[(s * E_ + e) * C_ + c] * t1L[c * V_ + v];
      sk += Wk[(s * E_ + e) * C_ + c] * t2L[c * V_ + v];
    }
    qL[idx] = sq;
    kL[idx] = sk;
  }
  __syncthreads();
  for (int idx = tid; idx < V_ * V_; idx += 128) {
    int u = idx / V_, v = idx - u * V_;
    float sum = 0.f;
#pragma unroll
    for (int e = 0; e < E_; ++e) sum += qL[e * V_ + v] * kL[e * V_ + u];
    lg[idx] = sum * (1.0f / (float)E_);
  }
  __syncthreads();
  _Float16* outp = A1T + (size_t)sn * 32 * 32;
  for (int idx = tid; idx < 32 * 32; idx += 128) {
    int u = idx >> 5, v = idx & 31;
    if (u >= V_ || v >= V_) outp[idx] = (_Float16)0.f;  // zero pad (K-pad exact)
  }
  if (tid < V_) {
    int u = tid;
    float m = -1e30f;
    for (int v = 0; v < V_; ++v) m = fmaxf(m, lg[u * V_ + v]);
    float den = 0.f;
    for (int v = 0; v < V_; ++v) den += __expf(lg[u * V_ + v] - m);
    float inv = 1.f / den;
    for (int v = 0; v < V_; ++v)
      outp[u * 32 + v] = (_Float16)(__expf(lg[u * V_ + v] - m) * inv);
  }
}

// ---------------- kernel 3: main WMMA kernel ----------------
// Block: 128 threads (4 waves). Tile: (n, 8 t-rows).
// GEMM1: P(64o x 32v) = Wd_s(64x64) @ X(64c x 32v)   [per t, per s]
// GEMM2: Y(64o x 32u) += P @ A1T_s(32v x 32u)
__global__ __launch_bounds__(128) void k_main(const float* __restrict__ x,
    const _Float16* __restrict__ WdH, const _Float16* __restrict__ A1T,
    const float* __restrict__ bnS, const float* __restrict__ bnB,
    const float* __restrict__ bdS, float* __restrict__ out) {
  __shared__ _Float16 XsT[256 * 72];     // [col=(tl*32+v)][c]  stride 72 halfs (144B)
  __shared__ _Float16 Pst[4 * 64 * 40];  // per-wave [o][v]     stride 40 halfs (80B)
  __shared__ _Float16 A1s[3 * 32 * 32];  // [s][u][v]
  __shared__ float bnSl[64], bnBl[64], bdSl[64];

  const int n = blockIdx.y;
  const int t0 = blockIdx.x * 8;
  const int tid = threadIdx.x;
  const int lane = tid & 31, wv = tid >> 5;
  const int lx = lane & 15, hi = lane >> 4;

  // cooperative load: x tile -> f16, transposed, v padded to 32, t padded
  for (int idx = tid; idx < C_ * 8 * 32; idx += 128) {
    int c = idx >> 8, rem = idx & 255, tl = rem >> 5, v = rem & 31;
    int t = t0 + tl;
    float val = (v < V_ && t < T_) ? x[((size_t)(n * C_ + c) * T_ + t) * V_ + v] : 0.f;
    XsT[(tl * 32 + v) * 72 + c] = (_Float16)val;
  }
  for (int idx = tid; idx < 3 * 32 * 32; idx += 128) {
    int s = idx >> 10, r = idx & 1023;
    A1s[idx] = A1T[((size_t)s * N_ + n) * 1024 + r];
  }
  if (tid < 64) { bnSl[tid] = bnS[tid]; bnBl[tid] = bnB[tid]; bdSl[tid] = bdS[tid]; }
  __syncthreads();

  _Float16* pw = &Pst[wv * 64 * 40];

  for (int tt = 0; tt < 2; ++tt) {
    const int tl = wv * 2 + tt;  // local t row handled by this wave
    v8f Y[4][2] = {};
#pragma unroll
    for (int s = 0; s < 3; ++s) {
      // ---- GEMM1 B operand: X columns for this t (K = c, contiguous) ----
      v16h Bt[2][2];
#pragma unroll
      for (int vt = 0; vt < 2; ++vt)
#pragma unroll
        for (int kb = 0; kb < 2; ++kb) {
          const _Float16* p = &XsT[(tl * 32 + vt * 16 + lx) * 72 + kb * 32 + hi * 16];
          Bt[vt][kb] = cat16(*(const v8h*)p, *(const v8h*)(p + 8));
        }
      v8f P[4][2] = {};
#pragma unroll
      for (int i = 0; i < 4; ++i) {
#pragma unroll
        for (int kb = 0; kb < 2; ++kb) {
          // A operand: Wd_s rows (L2-resident), halfs K in {0..15}/{16..31}
          const _Float16* g = &WdH[(size_t)(s * O_ + i * 16 + lx) * C_ + kb * 32 + hi * 16];
          v16h A = cat16(*(const v8h*)g, *(const v8h*)(g + 8));
#pragma unroll
          for (int vt = 0; vt < 2; ++vt)
            P[i][vt] = wmma_f16(A, Bt[vt][kb], P[i][vt]);
        }
      }
      // ---- stage P: C/D layout -> [o][v] f16 in per-wave LDS ----
#pragma unroll
      for (int i = 0; i < 4; ++i)
#pragma unroll
        for (int vt = 0; vt < 2; ++vt)
#pragma unroll
          for (int r = 0; r < 8; ++r)
            pw[(i * 16 + r + 8 * hi) * 40 + vt * 16 + lx] = (_Float16)P[i][vt][r];
      asm volatile("s_wait_dscnt 0" ::: "memory");  // same-wave LDS store->load ordering
      // ---- GEMM2: Y += P @ A1T_s  (K = 32 padded v, exact) ----
      v16h Bu[2];
#pragma unroll
      for (int ju = 0; ju < 2; ++ju) {
        const _Float16* p = &A1s[(s * 32 + ju * 16 + lx) * 32 + hi * 16];
        Bu[ju] = cat16(*(const v8h*)p, *(const v8h*)(p + 8));
      }
#pragma unroll
      for (int i = 0; i < 4; ++i) {
        const _Float16* pa = &pw[(i * 16 + lx) * 40 + hi * 16];
        v16h A2 = cat16(*(const v8h*)pa, *(const v8h*)(pa + 8));
#pragma unroll
        for (int ju = 0; ju < 2; ++ju)
          Y[i][ju] = wmma_f16(A2, Bu[ju], Y[i][ju]);
      }
    }
    // ---- finalize: bias + BN + residual + relu ----
    const int tg = t0 + tl;
    if (tg < T_) {
#pragma unroll
      for (int i = 0; i < 4; ++i)
#pragma unroll
        for (int ju = 0; ju < 2; ++ju) {
          const int u = ju * 16 + lx;
          if (u < V_) {
#pragma unroll
            for (int r = 0; r < 8; ++r) {
              const int o = i * 16 + r + 8 * hi;
              const size_t xo = ((size_t)(n * O_ + o) * T_ + tg) * V_ + u;
              float yb = (Y[i][ju][r] + bdSl[o]) * bnSl[o] + bnBl[o];
              float res = yb + x[xo];
              out[xo] = res > 0.f ? res : 0.f;
            }
          }
        }
    }
  }
}

extern "C" void kernel_launch(void* const* d_in, const int* in_sizes, int n_in,
                              void* d_out, int out_size, void* d_ws, size_t ws_size,
                              hipStream_t stream) {
  (void)in_sizes; (void)n_in; (void)out_size; (void)ws_size;
  const float* x     = (const float*)d_in[0];
  const float* Wt1   = (const float*)d_in[1];
  const float* bt1   = (const float*)d_in[2];
  const float* Wt2   = (const float*)d_in[3];
  const float* bt2   = (const float*)d_in[4];
  const float* Wq    = (const float*)d_in[5];
  const float* bq    = (const float*)d_in[6];
  const float* Wk    = (const float*)d_in[7];
  const float* bk    = (const float*)d_in[8];
  const float* Wd    = (const float*)d_in[9];
  const float* bd    = (const float*)d_in[10];
  const float* gamma = (const float*)d_in[11];
  const float* beta  = (const float*)d_in[12];
  const float* mean  = (const float*)d_in[13];
  const float* var   = (const float*)d_in[14];
  float* out = (float*)d_out;

  char* ws = (char*)d_ws;
  float*    t1o = (float*)(ws);                  // S*N*C*V f32 = 1,228,800 B
  float*    t2o = (float*)(ws + 1228800);        // 1,228,800 B
  _Float16* A1T = (_Float16*)(ws + 2457600);     // S*N*32*32 f16 = 393,216 B
  _Float16* WdH = (_Float16*)(ws + 2850816);     // S*O*C f16 = 24,576 B
  float*    bnS = (float*)(ws + 2875392);        // 64 f32
  float*    bnB = (float*)(ws + 2875392 + 256);  // 64 f32
  float*    bdS = (float*)(ws + 2875392 + 512);  // 64 f32

  k_prep<<<96, 128, 0, stream>>>(Wd, bd, gamma, beta, mean, var, WdH, bnS, bnB, bdS);
  k_tcollapse<<<4096, 32, 0, stream>>>(x, Wt1, bt1, Wt2, bt2, t1o, t2o);
  k_attn<<<192, 128, 0, stream>>>(t1o, t2o, Wq, bq, Wk, bk, A1T);
  k_main<<<dim3((T_ + 7) / 8, N_), 128, 0, stream>>>(x, WdH, A1T, bnS, bnB, bdS, out);
}
